// block_33844342292568
// MI455X (gfx1250) — matrix operand order repaired and checked by reading, not compiled, not run on hardware
//
#include <hip/hip_runtime.h>

// ---------------- problem constants ----------------
constexpr int Bc = 4;
constexpr int Tc = 2048;
constexpr int Cc = 1024;
constexpr int Hc = 16;
constexpr int HSc = 64;
constexpr int Mrows = Bc * Tc;            // 8192
constexpr float EPSc = 1e-5f;

typedef __bf16 bf16_t;
typedef __attribute__((ext_vector_type(16))) __bf16 v16bf;
typedef __attribute__((ext_vector_type(8)))  __bf16 bfx8;
typedef __attribute__((ext_vector_type(8)))  float  v8f;
typedef __attribute__((ext_vector_type(4)))  int    i32x4;
typedef __attribute__((ext_vector_type(4)))  unsigned int u32x4;
typedef __attribute__((ext_vector_type(8)))  unsigned int u32x8;

// AS-qualified pointee types for the async VMEM->LDS builtin
typedef __attribute__((address_space(1))) i32x4 as1_i32x4;
typedef __attribute__((address_space(3))) i32x4 as3_i32x4;

#ifndef __has_builtin
#define __has_builtin(x) 0
#endif
#if __has_builtin(__builtin_amdgcn_global_load_async_to_lds_b128)
#define HAVE_ASYNC_LDS 1
#else
#define HAVE_ASYNC_LDS 0
#endif

// ---------------- WMMA helpers ----------------
__device__ __forceinline__ v16bf make_v16(bfx8 lo, bfx8 hi) {
  union { bfx8 h[2]; v16bf v; } u;
  u.h[0] = lo; u.h[1] = hi;
  return u.v;
}

// A fragment (16x32 bf16, row-major, lda elements).
// Lane L: row L&15, k-chunks {k0..k0+7} and {k0+16..k0+23}, k0=(L>>4)*8.
__device__ __forceinline__ v16bf load_a_frag(const bf16_t* A, int lda, int lane) {
  int m  = lane & 15;
  int k0 = (lane >> 4) << 3;
  const bf16_t* p = A + (size_t)m * lda + k0;
  bfx8 lo = *(const bfx8*)(p);
  bfx8 hi = *(const bfx8*)(p + 16);
  return make_v16(lo, hi);
}

// B fragment (32x16 bf16) from N-major (K-contiguous) Bt[n][k].
// Lane L: col L&15, k-chunks {k0..k0+7} and {k0+16..k0+23}, k0=(L>>4)*8.
__device__ __forceinline__ v16bf load_b_frag(const bf16_t* Bt, int ldb, int lane) {
  int n  = lane & 15;
  int k0 = (lane >> 4) << 3;
  const bf16_t* p = Bt + (size_t)n * ldb + k0;
  bfx8 lo = *(const bfx8*)(p);
  bfx8 hi = *(const bfx8*)(p + 16);
  return make_v16(lo, hi);
}

__device__ __forceinline__ v8f wmma_bf16(v16bf a, v16bf b, v8f c) {
  return __builtin_amdgcn_wmma_f32_16x16x32_bf16(
      false, a, false, b, (short)0, c, false, false);
}

// ---------------- async copy global(16B) -> LDS ----------------
__device__ __forceinline__ void async_copy16(const bf16_t* g, bf16_t* l) {
#if HAVE_ASYNC_LDS
  __builtin_amdgcn_global_load_async_to_lds_b128(
      (as1_i32x4*)(void*)g, (as3_i32x4*)(void*)l, 0, 0);
#else
  *(bfx8*)l = *(const bfx8*)g;
#endif
}
__device__ __forceinline__ void async_wait_all() {
#if HAVE_ASYNC_LDS
  asm volatile("s_wait_asynccnt 0x0" ::: "memory");
#endif
}
__device__ __forceinline__ void wait_tensor(int pending) {
#if __has_builtin(__builtin_amdgcn_s_wait_tensorcnt)
  if (pending) __builtin_amdgcn_s_wait_tensorcnt(1);
  else         __builtin_amdgcn_s_wait_tensorcnt(0);
  asm volatile("" ::: "memory");
#else
  if (pending) asm volatile("s_wait_tensorcnt 0x1" ::: "memory");
  else         asm volatile("s_wait_tensorcnt 0x0" ::: "memory");
#endif
}

// ---------------- LayerNorm over TIME axis (ddof=1) ----------------
__global__ __launch_bounds__(256) void ln_time(
    const float* __restrict__ x, const float* __restrict__ g,
    const float* __restrict__ beta, bf16_t* __restrict__ out)
{
  __shared__ float sbuf[8][32];
  __shared__ float s2buf[8][32];
  int cl = threadIdx.x & 31;
  int ty = threadIdx.x >> 5;
  int c  = blockIdx.x * 32 + cl;
  size_t base = (size_t)blockIdx.y * Tc * Cc + c;
  float s = 0.f, s2 = 0.f;
  for (int t = ty; t < Tc; t += 8) {
    float v = x[base + (size_t)t * Cc];
    s += v; s2 += v * v;
  }
  sbuf[ty][cl] = s; s2buf[ty][cl] = s2;
  __syncthreads();
  float S = 0.f, S2 = 0.f;
  #pragma unroll
  for (int i = 0; i < 8; i++) { S += sbuf[i][cl]; S2 += s2buf[i][cl]; }
  float mean = S / (float)Tc;
  float var  = (S2 - S * mean) / (float)(Tc - 1);
  float inv  = rsqrtf(var + EPSc);
  float gg = g[c], bb = beta[c];
  for (int t = ty; t < Tc; t += 8) {
    float v = x[base + (size_t)t * Cc];
    out[base + (size_t)t * Cc] = (bf16_t)(gg * (v - mean) * inv + bb);
  }
}

// ---------------- weight prep ----------------
__global__ __launch_bounds__(256) void pack_qkv_w(
    const float* __restrict__ Wq, const float* __restrict__ Wk,
    const float* __restrict__ Wv, const float* __restrict__ bq,
    const float* __restrict__ bk, const float* __restrict__ bv,
    bf16_t* __restrict__ Wt, float* __restrict__ bias)
{
  int n = blockIdx.x;                 // 0 .. 3C-1
  int sel = n / Cc;
  int r = n - sel * Cc;
  int h = r / HSc, d = r - h * HSc;
  const float* W    = (sel == 0) ? Wq : (sel == 1) ? Wk : Wv;
  const float* bsrc = (sel == 0) ? bq : (sel == 1) ? bk : bv;
  for (int k = threadIdx.x; k < Cc; k += blockDim.x)
    Wt[(size_t)n * Cc + k] = (bf16_t)W[((size_t)h * Cc + k) * (size_t)HSc + d];
  if (threadIdx.x == 0) bias[n] = bsrc[r];
}

__global__ __launch_bounds__(256) void transpose_w(
    const float* __restrict__ W, bf16_t* __restrict__ Wt, int K, int N)
{
  int n = blockIdx.x;
  for (int k = threadIdx.x; k < K; k += blockDim.x)
    Wt[(size_t)n * K + k] = (bf16_t)W[(size_t)k * N + n];
}

// ---------------- LDS-tiled double-buffered WMMA GEMM ----------------
// C[M,N] = A[M,K] * Bt[N,K]^T (+bias) (+ReLU) (+resid). Block tile 64x128,
// K-step 64, async global->LDS copies, 8 waves = 4(M) x 2(N), wave tile 16x64.
constexpr int BM = 64, BN = 128, BK = 64;

template<int OUTF32, int RELU, int RES>
__global__ __launch_bounds__(256) void gemm_wmma(
    const bf16_t* __restrict__ A, const bf16_t* __restrict__ Bt,
    const float* __restrict__ bias, const float* __restrict__ resid,
    float* __restrict__ Cf, bf16_t* __restrict__ Cb,
    int M, int N, int K)
{
  __shared__ __align__(16) bf16_t As[2][BM * BK];
  __shared__ __align__(16) bf16_t Bs[2][BN * BK];
  const int tid  = threadIdx.x;
  const int lane = tid & 31;
  const int wv   = tid >> 5;
  const int mw   = wv >> 1;          // 0..3
  const int nw   = wv & 1;           // 0..1
  const int nblocks = N / BN;
  const int bm = blockIdx.x / nblocks;
  const int bn = blockIdx.x - bm * nblocks;
  const bf16_t* Ag = A  + (size_t)bm * BM * K;
  const bf16_t* Bg = Bt + (size_t)bn * BN * K;

  auto loadA = [&](int buf, int kt) {
    #pragma unroll
    for (int q = 0; q < 2; q++) {            // 512 16B-chunks / 256 threads
      int c = tid * 2 + q;
      int row = c >> 3, kc = c & 7;
      async_copy16(Ag + (size_t)row * K + kt * BK + kc * 8,
                   &As[buf][row * BK + kc * 8]);
    }
  };
  auto loadB = [&](int buf, int kt) {
    #pragma unroll
    for (int q = 0; q < 4; q++) {            // 1024 16B-chunks / 256 threads
      int c = tid * 4 + q;
      int row = c >> 3, kc = c & 7;
      async_copy16(Bg + (size_t)row * K + kt * BK + kc * 8,
                   &Bs[buf][row * BK + kc * 8]);
    }
  };

  v8f acc[4];
  #pragma unroll
  for (int j = 0; j < 4; j++)
    #pragma unroll
    for (int i = 0; i < 8; i++) acc[j][i] = 0.f;

  const int KT = K / BK;
  loadA(0, 0); loadB(0, 0);
  for (int kt = 0; kt < KT; kt++) {
    const int cur = kt & 1;
    async_wait_all();
    __syncthreads();                          // stage 'cur' visible to all waves
    if (kt + 1 < KT) { loadA(cur ^ 1, kt + 1); loadB(cur ^ 1, kt + 1); }
    const bf16_t* Ab = &As[cur][(mw * 16) * BK];
    const bf16_t* Bb = &Bs[cur][(nw * 64) * BK];
    #pragma unroll
    for (int kk = 0; kk < BK; kk += 32) {
      v16bf a = load_a_frag(Ab + kk, BK, lane);
      #pragma unroll
      for (int j = 0; j < 4; j++) {
        v16bf b = load_b_frag(Bb + (size_t)(j * 16) * BK + kk, BK, lane);
        acc[j] = wmma_bf16(a, b, acc[j]);
      }
    }
    __syncthreads();                          // done reading 'cur' before overwrite
  }

  const int n0 = bn * BN + nw * 64 + (lane & 15);
  const int m0 = bm * BM + mw * 16 + ((lane >> 4) << 3);
  #pragma unroll
  for (int j = 0; j < 4; j++) {
    int n = n0 + (j << 4);
    float bv = bias[n];
    #pragma unroll
    for (int i = 0; i < 8; i++) {
      int m = m0 + i;
      float v = acc[j][i] + bv;
      if (RELU) v = v > 0.f ? v : 0.f;
      if (RES)  v += resid[(size_t)m * N + n];
      if (OUTF32) Cf[(size_t)m * N + n] = v;
      else        Cb[(size_t)m * N + n] = (bf16_t)v;
    }
  }
}

// ---------------- flash attention with TDM K-tile staging ----------------
// 128 threads = 4 waves, one 16-query tile each. K tiles (32x64 bf16, row
// stride 3C) DMA'd to LDS by the Tensor Data Mover, double buffered per wave.
// LDS map (bytes): P[wv]   @ 0      + wv*1024   (16x32 bf16)
//                  Vt[wv]  @ 4096   + wv*4096   (64x32 bf16, transposed V)
//                  Kt[wv]  @ 20480  + wv*8192   (2 x 32x64 bf16)
constexpr unsigned P_BASE  = 0;
constexpr unsigned VT_BASE = 4096;
constexpr unsigned KT_BASE = 20480;

__global__ __launch_bounds__(128) void flash_attn(
    const bf16_t* __restrict__ qkv, bf16_t* __restrict__ attb)
{
  __shared__ __align__(16) char smem[KT_BASE + 4 * 8192];   // 53,248 B
  const int lane = threadIdx.x & 31;
  const int wv   = threadIdx.x >> 5;
  const int b = blockIdx.z, h = blockIdx.y;
  const int q0 = (blockIdx.x * 4 + wv) << 4;
  const int LD = 3 * Cc;
  const bf16_t* base = qkv + (size_t)b * Tc * LD;
  const bf16_t* qp = base + h * HSc;
  const bf16_t* kp = base + Cc + h * HSc;
  const bf16_t* vp = base + 2 * Cc + h * HSc;

  bf16_t* P  = (bf16_t*)(smem + P_BASE)  + wv * 512;
  bf16_t* Vt = (bf16_t*)(smem + VT_BASE) + wv * 2048;
  bf16_t* Kt = (bf16_t*)(smem + KT_BASE) + wv * 4096;
  const unsigned ktOff =
      (unsigned)__builtin_amdgcn_readfirstlane((int)(KT_BASE + wv * 8192));

  // issue one TDM 2D-tile load: 32 rows x 64 bf16, row stride LD elements
  auto issue_k = [&](int buf, int s0) {
    unsigned long long ga = (unsigned long long)(uintptr_t)(kp + (size_t)s0 * LD);
    unsigned galo = (unsigned)__builtin_amdgcn_readfirstlane((int)(ga & 0xffffffffu));
    unsigned gahi = (unsigned)__builtin_amdgcn_readfirstlane((int)(ga >> 32));
    u32x4 g0;
    g0[0] = 1u;                                   // count=1, user descriptor
    g0[1] = ktOff + (unsigned)(buf * 4096);       // lds_addr (bytes)
    g0[2] = galo;                                 // global_addr[31:0]
    g0[3] = (gahi & 0x01ffffffu) | (2u << 30);    // global_addr[56:32] | type=2
    u32x8 g1;
    g1[0] = 0x00010000u;                          // data_size = 2 bytes
    g1[1] = 64u << 16;                            // tensor_dim0 = 64
    g1[2] = (unsigned)Tc << 16;                   // tensor_dim1[15:0] = 2048
    g1[3] = (64u << 16) | ((unsigned)Tc >> 16);   // tile_dim0=64 | tensor_dim1 hi
    g1[4] = 32u;                                  // tile_dim1 = 32
    g1[5] = (unsigned)LD;                         // tensor_dim0_stride = 3072
    g1[6] = 0u;
    g1[7] = 0u;
    asm volatile("tensor_load_to_lds %0, %1" :: "s"(g0), "s"(g1) : "memory");
  };

  v16bf qa0 = load_a_frag(qp + (size_t)q0 * LD, LD, lane);
  v16bf qa1 = load_a_frag(qp + (size_t)q0 * LD + 32, LD, lane);

  const float scale = 0.03125f;   // 1024^-0.5
  const int rbase = (lane >> 4) << 3;
  const int cidx  = lane & 15;

  float mrun[8], lrun[8];
  v8f o[4];
  #pragma unroll
  for (int i = 0; i < 8; i++) { mrun[i] = -3.0e38f; lrun[i] = 0.f; }
  #pragma unroll
  for (int j = 0; j < 4; j++)
    #pragma unroll
    for (int i = 0; i < 8; i++) o[j][i] = 0.f;

  const int nIter = (q0 + 16 + 31) >> 5;          // #32-key tiles
  issue_k(0, 0);
  for (int it = 0; it < nIter; it++) {
    const int s0  = it << 5;
    const int cur = it & 1;
    const bool more = (it + 1 < nIter);
    if (more) issue_k(cur ^ 1, s0 + 32);
    wait_tensor(more ? 1 : 0);                    // current K tile resident

    const bf16_t* Kb = Kt + cur * 2048;           // 32x64 row-major in LDS
    v8f sc[2];
    #pragma unroll
    for (int sub = 0; sub < 2; sub++) {
      #pragma unroll
      for (int i = 0; i < 8; i++) sc[sub][i] = 0.f;
      v16bf b0 = load_b_frag(Kb + (sub * 16) * 64,      64, lane);
      v16bf b1 = load_b_frag(Kb + (sub * 16) * 64 + 32, 64, lane);
      sc[sub] = wmma_bf16(qa0, b0, sc[sub]);
      sc[sub] = wmma_bf16(qa1, b1, sc[sub]);
    }
    // scale + causal mask, row max
    float rmax[8];
    #pragma unroll
    for (int i = 0; i < 8; i++) {
      int m = q0 + rbase + i;
      #pragma unroll
      for (int sub = 0; sub < 2; sub++) {
        int n = s0 + sub * 16 + cidx;
        float v = sc[sub][i] * scale;
        sc[sub][i] = (n <= m) ? v : -3.0e38f;
      }
      rmax[i] = fmaxf(sc[0][i], sc[1][i]);
    }
    #pragma unroll
    for (int off = 1; off < 16; off <<= 1)
      #pragma unroll
      for (int i = 0; i < 8; i++)
        rmax[i] = fmaxf(rmax[i], __shfl_xor(rmax[i], off, 32));
    // online softmax update
    float alpha[8], rsum[8];
    #pragma unroll
    for (int i = 0; i < 8; i++) {
      float mn = fmaxf(mrun[i], rmax[i]);
      alpha[i] = __expf(mrun[i] - mn);
      mrun[i]  = mn;
      float p0 = __expf(sc[0][i] - mn);
      float p1 = __expf(sc[1][i] - mn);
      sc[0][i] = p0; sc[1][i] = p1;
      rsum[i] = p0 + p1;
    }
    #pragma unroll
    for (int off = 1; off < 16; off <<= 1)
      #pragma unroll
      for (int i = 0; i < 8; i++)
        rsum[i] += __shfl_xor(rsum[i], off, 32);
    #pragma unroll
    for (int i = 0; i < 8; i++) lrun[i] = lrun[i] * alpha[i] + rsum[i];
    #pragma unroll
    for (int j = 0; j < 4; j++)
      #pragma unroll
      for (int i = 0; i < 8; i++) o[j][i] *= alpha[i];
    // P -> LDS (16x32 row-major) to re-layout into an A fragment
    #pragma unroll
    for (int i = 0; i < 8; i++)
      #pragma unroll
      for (int sub = 0; sub < 2; sub++)
        P[(rbase + i) * 32 + sub * 16 + cidx] = (bf16_t)sc[sub][i];
    // stage V transposed: Vt[d][key]
    {
      const bf16_t* vrow = vp + (size_t)(s0 + lane) * LD;
      #pragma unroll
      for (int d = 0; d < 64; d++) Vt[d * 32 + lane] = vrow[d];
    }
    asm volatile("s_wait_dscnt 0x0" ::: "memory");
    // O += P V
    v16bf pa = load_a_frag(P, 32, lane);
    #pragma unroll
    for (int j = 0; j < 4; j++) {
      v16bf vb = load_b_frag(Vt + (j * 16) * 32, 32, lane);
      o[j] = wmma_bf16(pa, vb, o[j]);
    }
    asm volatile("s_wait_dscnt 0x0" ::: "memory");
  }
  // normalize + store concat-head layout [B,T,C]
  #pragma unroll
  for (int j = 0; j < 4; j++) {
    #pragma unroll
    for (int i = 0; i < 8; i++) {
      float v = o[j][i] / lrun[i];
      size_t idx = ((size_t)b * Tc + q0 + rbase + i) * Cc + h * HSc + j * 16 + cidx;
      attb[idx] = (bf16_t)v;
    }
  }
}

// ---------------- launcher ----------------
extern "C" void kernel_launch(void* const* d_in, const int* in_sizes, int n_in,
                              void* d_out, int out_size, void* d_ws, size_t ws_size,
                              hipStream_t stream)
{
  const float* x    = (const float*)d_in[0];
  const float* ln1g = (const float*)d_in[1];
  const float* ln1b = (const float*)d_in[2];
  const float* ln2g = (const float*)d_in[3];
  const float* ln2b = (const float*)d_in[4];
  const float* Wq   = (const float*)d_in[5];
  const float* bq   = (const float*)d_in[6];
  const float* Wk   = (const float*)d_in[7];
  const float* bk   = (const float*)d_in[8];
  const float* Wv   = (const float*)d_in[9];
  const float* bv   = (const float*)d_in[10];
  const float* Wo   = (const float*)d_in[11];
  const float* bo   = (const float*)d_in[12];
  const float* W1   = (const float*)d_in[13];
  const float* b1   = (const float*)d_in[14];
  const float* W2   = (const float*)d_in[15];
  const float* b2   = (const float*)d_in[16];

  const size_t BTC = (size_t)Mrows * Cc;
  char* w = (char*)d_ws;
  auto grab = [&](size_t bytes) { char* p = w; w += (bytes + 255) & ~(size_t)255; return p; };

  bf16_t* hb   = (bf16_t*)grab(BTC * 2);
  bf16_t* qkvb = (bf16_t*)grab(BTC * 3 * 2);
  bf16_t* attb = (bf16_t*)grab(BTC * 2);
  float*  x2   = (float*) grab(BTC * 4);
  bf16_t* h2b  = (bf16_t*)grab(BTC * 2);
  bf16_t* ff1  = (bf16_t*)grab(BTC * 4 * 2);
  bf16_t* wqkv = (bf16_t*)grab((size_t)3 * Cc * Cc * 2);
  bf16_t* wot  = (bf16_t*)grab((size_t)Cc * Cc * 2);
  bf16_t* w1t  = (bf16_t*)grab((size_t)4 * Cc * Cc * 2);
  bf16_t* w2t  = (bf16_t*)grab((size_t)4 * Cc * Cc * 2);
  float*  bqkv = (float*) grab((size_t)3 * Cc * 4);

  dim3 lnGrid(Cc / 32, Bc);
  ln_time<<<lnGrid, 256, 0, stream>>>(x, ln1g, ln1b, hb);

  pack_qkv_w<<<3 * Cc, 256, 0, stream>>>(Wq, Wk, Wv, bq, bk, bv, wqkv, bqkv);
  transpose_w<<<Cc,     256, 0, stream>>>(Wo, wot, Cc,     Cc);
  transpose_w<<<4 * Cc, 256, 0, stream>>>(W1, w1t, Cc,     4 * Cc);
  transpose_w<<<Cc,     256, 0, stream>>>(W2, w2t, 4 * Cc, Cc);

  // QKV projection: [8192,1024] x [1024,3072]
  gemm_wmma<0, 0, 0><<<(Mrows / BM) * (3 * Cc / BN), 256, 0, stream>>>(
      hb, wqkv, bqkv, nullptr, nullptr, qkvb, Mrows, 3 * Cc, Cc);

  flash_attn<<<dim3(Tc / 64, Hc, Bc), 128, 0, stream>>>(qkvb, attb);

  // out projection + residual: x2 = x + att@Wo + bo
  gemm_wmma<1, 0, 1><<<(Mrows / BM) * (Cc / BN), 256, 0, stream>>>(
      attb, wot, bo, x, x2, nullptr, Mrows, Cc, Cc);

  ln_time<<<lnGrid, 256, 0, stream>>>(x2, ln2g, ln2b, h2b);

  // FFN up + ReLU
  gemm_wmma<0, 1, 0><<<(Mrows / BM) * (4 * Cc / BN), 256, 0, stream>>>(
      h2b, w1t, b1, nullptr, nullptr, ff1, Mrows, 4 * Cc, Cc);

  // FFN down + residual -> d_out
  gemm_wmma<1, 0, 1><<<(Mrows / BM) * (Cc / BN), 256, 0, stream>>>(
      ff1, w2t, b2, x2, (float*)d_out, nullptr, Mrows, Cc, 4 * Cc);
}
